// PairEnergies_87514253623875
// MI455X (gfx1250) — matrix operand order repaired and checked by reading, not compiled, not run on hardware
//
#include <hip/hip_runtime.h>

// Problem constants (match reference)
#define BB   8
#define NN   1024
#define KKN  30
#define HH   128
#define LL   3
#define OUTD 400
#define NKE  (NN*KKN)        // 30720 edges per batch
#define ME   (BB*NKE)        // 245760 total edges
#define MTE  (ME/16)         // 15360 edge tiles
#define MV   (BB*NN)         // 8192 nodes
#define MTV  (MV/16)         // 512 node tiles
#define SCALE_INV (1.0f/30.0f)

typedef __attribute__((ext_vector_type(16))) _Float16 v16h;
typedef __attribute__((ext_vector_type(8)))  _Float16 v8h;
typedef __attribute__((ext_vector_type(8)))  float    v8f;

union AFrag { v16h v; v8h h[2]; };

#define WMMA_F16(A, Bm, C) \
  __builtin_amdgcn_wmma_f32_16x16x32_f16(false, (A), false, (Bm), (short)0, (C), false, false)

__device__ __forceinline__ float wave_sum32(float v) {
#pragma unroll
  for (int o = 16; o > 0; o >>= 1) v += __shfl_xor(v, o, 32);
  return v;
}

// ---------------------------------------------------------------- elementwise
__global__ void k_cvt(const float* __restrict__ s, _Float16* __restrict__ d, int n) {
  int i = blockIdx.x * blockDim.x + threadIdx.x;
  if (i < n) d[i] = (_Float16)s[i];
}

__global__ void k_zero(float* __restrict__ p, int n) {
  int i = blockIdx.x * blockDim.x + threadIdx.x;
  if (i < n) p[i] = 0.0f;
}

// --------------------------------------------------- C[M,128] = A[M,128]@W + b
// A f16 row-major, W f16 row-major [128,128]. Writes f32 and f16 copies.
__global__ void __launch_bounds__(256)
k_gemm128(const _Float16* __restrict__ A, const _Float16* __restrict__ W,
          const float* __restrict__ bias, float* __restrict__ Cf,
          _Float16* __restrict__ Ch, int mtiles) {
  int lane = threadIdx.x & 31, wid = threadIdx.x >> 5, nw = blockDim.x >> 5;
  int m = lane & 15, g = lane >> 4, colb = lane & 15;
  for (int tile = blockIdx.x * nw + wid; tile < mtiles; tile += gridDim.x * nw) {
    const _Float16* arow = A + ((long)tile * 16 + m) * HH;
    v8f zero = {};
    v8f acc[8];
#pragma unroll
    for (int t = 0; t < 8; ++t) acc[t] = zero;
#pragma unroll
    for (int c = 0; c < 4; ++c) {
      AFrag a;
      a.h[0] = *(const v8h*)(arow + c * 32 + g * 8);
      a.h[1] = *(const v8h*)(arow + c * 32 + 16 + g * 8);
#pragma unroll
      for (int t = 0; t < 8; ++t) {
        v16h bf;
        for (int s = 0; s < 16; ++s) bf[s] = W[(long)(c * 32 + (s < 8 ? g * 8 + s : 16 + g * 8 + s - 8)) * HH + t * 16 + colb];
        acc[t] = WMMA_F16(a.v, bf, acc[t]);
      }
    }
#pragma unroll
    for (int t = 0; t < 8; ++t) {
      float bb = bias[t * 16 + colb];
#pragma unroll
      for (int r = 0; r < 8; ++r) {
        long off = ((long)tile * 16 + r + 8 * g) * HH + t * 16 + colb;
        float v = acc[t][r] + bb;
        Cf[off] = v;
        Ch[off] = (_Float16)v;
      }
    }
  }
}

// ------------------------------------------- fused 3-layer MLP on edge tiles
// X = [h_i | h_j | h_E] (16x384 f16, gathered directly from global),
// H1 = relu(X@W1+b1); H2 = relu(H1@W2+b2); M = H2@W3+b3
// MODE 0 (edge): out = resid + mask*M ; MODE 1 (node): out = mask*M
template <int MODE>
__global__ void __launch_bounds__(256)
k_mlp3(const _Float16* __restrict__ hVh, const _Float16* __restrict__ hEh,
       const int* __restrict__ Eidx, const float* __restrict__ xmask,
       const _Float16* __restrict__ W1, const float* __restrict__ b1,
       const _Float16* __restrict__ W2, const float* __restrict__ b2,
       const _Float16* __restrict__ W3, const float* __restrict__ b3,
       const float* __restrict__ resid, float* __restrict__ outG) {
  extern __shared__ _Float16 smem[];
  int lane = threadIdx.x & 31, wid = threadIdx.x >> 5, nw = blockDim.x >> 5;
  _Float16* mid = smem + wid * (16 * HH);   // 4KB per wave
  int m = lane & 15, g = lane >> 4, colb = lane & 15;
  __builtin_prefetch(W2, 0, 0);
  __builtin_prefetch(W3, 0, 0);
  for (int tile = blockIdx.x * nw + wid; tile < MTE; tile += gridDim.x * nw) {
    long edge = (long)tile * 16 + m;
    int b = (int)(edge / NKE);
    int rem = (int)(edge % NKE);
    int n = rem / KKN;
    int j = Eidx[edge];
    const _Float16* rowI = hVh + ((long)b * NN + n) * HH;
    const _Float16* rowJ = hVh + ((long)b * NN + j) * HH;
    const _Float16* rowE = hEh + edge * HH;
    float msk = xmask[b * NN + n] * xmask[b * NN + j];

    v8f zero = {};
    v8f acc[8];
    // ---- GEMM1: 16x384 @ 384x128
#pragma unroll
    for (int t = 0; t < 8; ++t) acc[t] = zero;
#pragma unroll
    for (int c = 0; c < 12; ++c) {
      const _Float16* src = (c < 4) ? rowI : (c < 8) ? rowJ : rowE;
      int off = (c & 3) * 32 + g * 8;
      AFrag a;
      a.h[0] = *(const v8h*)(src + off);
      a.h[1] = *(const v8h*)(src + off + 16);
#pragma unroll
      for (int t = 0; t < 8; ++t) {
        v16h bf;
        for (int s = 0; s < 16; ++s) bf[s] = W1[(long)(c * 32 + (s < 8 ? g * 8 + s : 16 + g * 8 + s - 8)) * HH + t * 16 + colb];
        acc[t] = WMMA_F16(a.v, bf, acc[t]);
      }
    }
#pragma unroll
    for (int t = 0; t < 8; ++t) {
      float bb = b1[t * 16 + colb];
#pragma unroll
      for (int r = 0; r < 8; ++r) {
        float v = acc[t][r] + bb;
        mid[(r + 8 * g) * HH + t * 16 + colb] = (_Float16)(v > 0.f ? v : 0.f);
      }
    }
    asm volatile("s_wait_dscnt 0" ::: "memory");

    // ---- GEMM2: 16x128 @ 128x128
#pragma unroll
    for (int t = 0; t < 8; ++t) acc[t] = zero;
#pragma unroll
    for (int c = 0; c < 4; ++c) {
      AFrag a;
      a.h[0] = *(const v8h*)(mid + m * HH + c * 32 + g * 8);
      a.h[1] = *(const v8h*)(mid + m * HH + c * 32 + 16 + g * 8);
#pragma unroll
      for (int t = 0; t < 8; ++t) {
        v16h bf;
        for (int s = 0; s < 16; ++s) bf[s] = W2[(long)(c * 32 + (s < 8 ? g * 8 + s : 16 + g * 8 + s - 8)) * HH + t * 16 + colb];
        acc[t] = WMMA_F16(a.v, bf, acc[t]);
      }
    }
#pragma unroll
    for (int t = 0; t < 8; ++t) {
      float bb = b2[t * 16 + colb];
#pragma unroll
      for (int r = 0; r < 8; ++r) {
        float v = acc[t][r] + bb;
        mid[(r + 8 * g) * HH + t * 16 + colb] = (_Float16)(v > 0.f ? v : 0.f);
      }
    }
    asm volatile("s_wait_dscnt 0" ::: "memory");

    // ---- GEMM3: 16x128 @ 128x128
#pragma unroll
    for (int t = 0; t < 8; ++t) acc[t] = zero;
#pragma unroll
    for (int c = 0; c < 4; ++c) {
      AFrag a;
      a.h[0] = *(const v8h*)(mid + m * HH + c * 32 + g * 8);
      a.h[1] = *(const v8h*)(mid + m * HH + c * 32 + 16 + g * 8);
#pragma unroll
      for (int t = 0; t < 8; ++t) {
        v16h bf;
        for (int s = 0; s < 16; ++s) bf[s] = W3[(long)(c * 32 + (s < 8 ? g * 8 + s : 16 + g * 8 + s - 8)) * HH + t * 16 + colb];
        acc[t] = WMMA_F16(a.v, bf, acc[t]);
      }
    }
    float maskr[8];
#pragma unroll
    for (int r = 0; r < 8; ++r) maskr[r] = __shfl(msk, r + 8 * g, 32);
#pragma unroll
    for (int t = 0; t < 8; ++t) {
      float bb = b3[t * 16 + colb];
#pragma unroll
      for (int r = 0; r < 8; ++r) {
        long off = ((long)tile * 16 + r + 8 * g) * HH + t * 16 + colb;
        float v = acc[t][r] + bb;
        if (MODE == 0) outG[off] = resid[off] + maskr[r] * v;
        else           outG[off] = maskr[r] * v;
      }
    }
  }
}

// ------------------------------------------ per-edge LayerNorm(pre) -> hE f32
__global__ void __launch_bounds__(256)
k_ln_edge(const float* __restrict__ pre, const float* __restrict__ gam,
          const float* __restrict__ bet, float* __restrict__ hE) {
  int lane = threadIdx.x & 31, wid = threadIdx.x >> 5, nw = blockDim.x >> 5;
  for (int row = blockIdx.x * nw + wid; row < ME; row += gridDim.x * nw) {
    float4 v = *(const float4*)(pre + (long)row * HH + lane * 4);
    float mean = wave_sum32(v.x + v.y + v.z + v.w) * (1.0f / HH);
    float dx = v.x - mean, dy = v.y - mean, dz = v.z - mean, dw = v.w - mean;
    float var = wave_sum32(dx * dx + dy * dy + dz * dz + dw * dw) * (1.0f / HH);
    float rs = rsqrtf(var + 1e-5f);
    int h0 = lane * 4;
    float* o = hE + (long)row * HH + h0;
    o[0] = dx * rs * gam[h0 + 0] + bet[h0 + 0];
    o[1] = dy * rs * gam[h0 + 1] + bet[h0 + 1];
    o[2] = dz * rs * gam[h0 + 2] + bet[h0 + 2];
    o[3] = dw * rs * gam[h0 + 3] + bet[h0 + 3];
  }
}

// ------------------------------------------------- merge_dups (scatter-mean)
__global__ void k_scatter(const float* __restrict__ hE, const int* __restrict__ inv,
                          float* __restrict__ seg, float* __restrict__ cnt) {
  long idx = (long)blockIdx.x * blockDim.x + threadIdx.x;
  if (idx >= (long)ME * HH) return;
  int edge = (int)(idx >> 7), h = (int)(idx & 127);
  int b = edge / NKE;
  int s = inv[edge];
  atomicAdd(seg + ((long)b * NKE + s) * HH + h, hE[idx]);
  if (h == 0) atomicAdd(cnt + (long)b * NKE + s, 1.0f);
}

__global__ void k_merge(const float* __restrict__ seg, const float* __restrict__ cnt,
                        const int* __restrict__ inv, const float* __restrict__ xmask,
                        float* __restrict__ hE, _Float16* __restrict__ hEh) {
  long idx = (long)blockIdx.x * blockDim.x + threadIdx.x;
  if (idx >= (long)ME * HH) return;
  int edge = (int)(idx >> 7), h = (int)(idx & 127);
  int b = edge / NKE;
  int rem = edge % NKE;
  int n = rem / KKN;
  int s = inv[edge];
  float c = cnt[(long)b * NKE + s];
  float val = seg[((long)b * NKE + s) * HH + h] / fmaxf(c, 1.0f) * xmask[b * NN + n];
  hE[idx] = val;
  hEh[idx] = (_Float16)val;
}

// ------------------------ node update: dh = sum_k msg /30 ; LN ; mask ; f16
__global__ void __launch_bounds__(256)
k_node_ln(const float* __restrict__ msg, float* __restrict__ hV,
          _Float16* __restrict__ hVh, const float* __restrict__ gam,
          const float* __restrict__ bet, const float* __restrict__ xmask) {
  int lane = threadIdx.x & 31, wid = threadIdx.x >> 5, nw = blockDim.x >> 5;
  for (int node = blockIdx.x * nw + wid; node < MV; node += gridDim.x * nw) {
    float sx = 0, sy = 0, sz = 0, sw = 0;
    const float* base = msg + (long)node * KKN * HH + lane * 4;
    for (int k = 0; k < KKN; ++k) {
      float4 v = *(const float4*)(base + (long)k * HH);
      sx += v.x; sy += v.y; sz += v.z; sw += v.w;
    }
    float4 hv = *(const float4*)(hV + (long)node * HH + lane * 4);
    float px = hv.x + sx * SCALE_INV, py = hv.y + sy * SCALE_INV;
    float pz = hv.z + sz * SCALE_INV, pw = hv.w + sw * SCALE_INV;
    float mean = wave_sum32(px + py + pz + pw) * (1.0f / HH);
    float dx = px - mean, dy = py - mean, dz = pz - mean, dw = pw - mean;
    float var = wave_sum32(dx * dx + dy * dy + dz * dz + dw * dw) * (1.0f / HH);
    float rs = rsqrtf(var + 1e-5f);
    float xm = xmask[node];
    int h0 = lane * 4;
    float o0 = (dx * rs * gam[h0 + 0] + bet[h0 + 0]) * xm;
    float o1 = (dy * rs * gam[h0 + 1] + bet[h0 + 1]) * xm;
    float o2 = (dz * rs * gam[h0 + 2] + bet[h0 + 2]) * xm;
    float o3 = (dw * rs * gam[h0 + 3] + bet[h0 + 3]) * xm;
    float* o = hV + (long)node * HH + h0;
    o[0] = o0; o[1] = o1; o[2] = o2; o[3] = o3;
    _Float16* oh = hVh + (long)node * HH + h0;
    oh[0] = (_Float16)o0; oh[1] = (_Float16)o1;
    oh[2] = (_Float16)o2; oh[3] = (_Float16)o3;
  }
}

// --------------------------------- output head: etab = hE @ Wout + bout (400)
__global__ void __launch_bounds__(256)
k_out(const _Float16* __restrict__ hEh, const _Float16* __restrict__ Wo,
      const float* __restrict__ bo, float* __restrict__ out) {
  int lane = threadIdx.x & 31, wid = threadIdx.x >> 5, nw = blockDim.x >> 5;
  int m = lane & 15, g = lane >> 4, colb = lane & 15;
  const int NJOBS = MTE * 5;  // 25 n-tiles split into 5 jobs of 5
  for (int job = blockIdx.x * nw + wid; job < NJOBS; job += gridDim.x * nw) {
    int tile = job / 5;
    int t0 = (job % 5) * 5;
    const _Float16* arow = hEh + ((long)tile * 16 + m) * HH;
    v8f zero = {};
    v8f acc[5];
#pragma unroll
    for (int t = 0; t < 5; ++t) acc[t] = zero;
#pragma unroll
    for (int c = 0; c < 4; ++c) {
      AFrag a;
      a.h[0] = *(const v8h*)(arow + c * 32 + g * 8);
      a.h[1] = *(const v8h*)(arow + c * 32 + 16 + g * 8);
#pragma unroll
      for (int t = 0; t < 5; ++t) {
        v16h bf;
        for (int s = 0; s < 16; ++s) bf[s] = Wo[(long)(c * 32 + (s < 8 ? g * 8 + s : 16 + g * 8 + s - 8)) * OUTD + (t0 + t) * 16 + colb];
        acc[t] = WMMA_F16(a.v, bf, acc[t]);
      }
    }
#pragma unroll
    for (int t = 0; t < 5; ++t) {
      int col = (t0 + t) * 16 + colb;
      float bb = bo[col];
#pragma unroll
      for (int r = 0; r < 8; ++r) {
        out[((long)tile * 16 + r + 8 * g) * OUTD + col] = acc[t][r] + bb;
      }
    }
  }
}

// ============================================================================
extern "C" void kernel_launch(void* const* d_in, const int* in_sizes, int n_in,
                              void* d_out, int out_size, void* d_ws, size_t ws_size,
                              hipStream_t stream) {
  (void)in_sizes; (void)n_in; (void)out_size; (void)ws_size;
  const float* V     = (const float*)d_in[0];
  const float* E     = (const float*)d_in[1];
  const float* xmask = (const float*)d_in[2];
  const int*   Eidx  = (const int*)d_in[3];
  const int*   inv   = (const int*)d_in[4];
  const float* Wv    = (const float*)d_in[5];
  const float* bv    = (const float*)d_in[6];
  const float* We    = (const float*)d_in[7];
  const float* be    = (const float*)d_in[8];
  const float* eW1   = (const float*)d_in[9];
  const float* eb1   = (const float*)d_in[10];
  const float* eW2   = (const float*)d_in[11];
  const float* eb2   = (const float*)d_in[12];
  const float* eW3   = (const float*)d_in[13];
  const float* eb3   = (const float*)d_in[14];
  const float* eg    = (const float*)d_in[15];
  const float* ebt   = (const float*)d_in[16];
  const float* nW1   = (const float*)d_in[17];
  const float* nb1   = (const float*)d_in[18];
  const float* nW2   = (const float*)d_in[19];
  const float* nb2   = (const float*)d_in[20];
  const float* nW3   = (const float*)d_in[21];
  const float* nb3   = (const float*)d_in[22];
  const float* ng    = (const float*)d_in[23];
  const float* nbt   = (const float*)d_in[24];
  const float* Wout  = (const float*)d_in[25];
  const float* bout  = (const float*)d_in[26];

  // workspace layout
  size_t off = 0;
  char* base = (char*)d_ws;
  auto wsalloc = [&](size_t bytes) -> void* {
    void* p = base + off;
    off += (bytes + 255) & ~(size_t)255;
    return p;
  };
  float*    hV   = (float*)wsalloc((size_t)MV * HH * 4);
  _Float16* hVh  = (_Float16*)wsalloc((size_t)MV * HH * 2);
  float*    hE   = (float*)wsalloc((size_t)ME * HH * 4);
  _Float16* hEh  = (_Float16*)wsalloc((size_t)ME * HH * 2);
  float*    tmp  = (float*)wsalloc((size_t)ME * HH * 4);   // msg / pre-LN / seg sums
  float*    cnt  = (float*)wsalloc((size_t)BB * NKE * 4);
  _Float16* Wvh  = (_Float16*)wsalloc((size_t)HH * HH * 2);
  _Float16* Weh  = (_Float16*)wsalloc((size_t)HH * HH * 2);
  _Float16* eW1h = (_Float16*)wsalloc((size_t)LL * 3 * HH * HH * 2);
  _Float16* eW2h = (_Float16*)wsalloc((size_t)LL * HH * HH * 2);
  _Float16* eW3h = (_Float16*)wsalloc((size_t)LL * HH * HH * 2);
  _Float16* nW1h = (_Float16*)wsalloc((size_t)LL * 3 * HH * HH * 2);
  _Float16* nW2h = (_Float16*)wsalloc((size_t)LL * HH * HH * 2);
  _Float16* nW3h = (_Float16*)wsalloc((size_t)LL * HH * HH * 2);
  _Float16* Wouth = (_Float16*)wsalloc((size_t)HH * OUTD * 2);

  auto cvt = [&](const float* s, _Float16* d, int n) {
    k_cvt<<<(n + 255) / 256, 256, 0, stream>>>(s, d, n);
  };
  // weights -> f16 (row-major == B-fragment friendly)
  cvt(Wv, Wvh, HH * HH);
  cvt(We, Weh, HH * HH);
  cvt(eW1, eW1h, LL * 3 * HH * HH);
  cvt(eW2, eW2h, LL * HH * HH);
  cvt(eW3, eW3h, LL * HH * HH);
  cvt(nW1, nW1h, LL * 3 * HH * HH);
  cvt(nW2, nW2h, LL * HH * HH);
  cvt(nW3, nW3h, LL * HH * HH);
  cvt(Wout, Wouth, HH * OUTD);

  // input activations -> f16 (staged inside tmp, which is free right now)
  _Float16* EhIn = (_Float16*)tmp;
  _Float16* VhIn = (_Float16*)tmp + (size_t)ME * HH;  // fits: 63MB + 2MB < 126MB
  cvt(E, EhIn, ME * HH);
  cvt(V, VhIn, MV * HH);

  // initial projections
  k_gemm128<<<64, 256, 0, stream>>>(VhIn, Wvh, bv, hV, hVh, MTV);
  k_gemm128<<<480, 256, 0, stream>>>(EhIn, Weh, be, hE, hEh, MTE);

  const int NELEM = ME * HH;  // 31,457,280
  for (int l = 0; l < LL; ++l) {
    // edge message + residual
    k_mlp3<0><<<480, 256, 32768, stream>>>(
        hVh, hEh, Eidx, xmask,
        eW1h + (size_t)l * 3 * HH * HH, eb1 + l * HH,
        eW2h + (size_t)l * HH * HH, eb2 + l * HH,
        eW3h + (size_t)l * HH * HH, eb3 + l * HH,
        hE, tmp);
    // layernorm -> hE (f32)
    k_ln_edge<<<960, 256, 0, stream>>>(tmp, eg + l * HH, ebt + l * HH, hE);
    // merge duplicate (undirected) edges: scatter-mean via atomics into tmp
    k_zero<<<(NELEM + 255) / 256, 256, 0, stream>>>(tmp, NELEM);
    k_zero<<<(BB * NKE + 255) / 256, 256, 0, stream>>>(cnt, BB * NKE);
    k_scatter<<<(NELEM + 255) / 256, 256, 0, stream>>>(hE, inv, tmp, cnt);
    k_merge<<<(NELEM + 255) / 256, 256, 0, stream>>>(tmp, cnt, inv, xmask, hE, hEh);
    // node message (uses old h_V, new h_E)
    k_mlp3<1><<<480, 256, 32768, stream>>>(
        hVh, hEh, Eidx, xmask,
        nW1h + (size_t)l * 3 * HH * HH, nb1 + l * HH,
        nW2h + (size_t)l * HH * HH, nb2 + l * HH,
        nW3h + (size_t)l * HH * HH, nb3 + l * HH,
        hE /*unused*/, tmp);
    // node reduce + layernorm + mask
    k_node_ln<<<128, 256, 0, stream>>>(tmp, hV, hVh, ng + l * HH, nbt + l * HH, xmask);
  }

  // output head
  k_out<<<960, 256, 0, stream>>>(hEh, Wouth, bout, (float*)d_out);
}
